// WindowAttention_29351806501373
// MI455X (gfx1250) — matrix operand order repaired and checked by reading, not compiled, not run on hardware
//
#include <hip/hip_runtime.h>

typedef __attribute__((ext_vector_type(16))) _Float16 v16h;
typedef __attribute__((ext_vector_type(8)))  float    v8f;

#define WSZ   14
#define NH    12
#define HD    64
#define CDIM  768
#define NTOK  196
#define NWIN  256            // 16 images * 4*4 windows
#define QPAD  208            // 13 * 16
#define KPAD  224            // 7  * 32

// CDNA5 async global->LDS DMA (ASYNCcnt path), guarded for toolchain support.
#if defined(__gfx1250__) && __has_builtin(__builtin_amdgcn_global_load_async_to_lds_b128)
#define HAVE_ASYNC_LDS 1
#else
#define HAVE_ASYNC_LDS 0
#endif

#if HAVE_ASYNC_LDS
// The builtin takes (int4 addrspace(1)*, int4 addrspace(3)*, imm offset, imm cpol)
// per the clang prototype.  Go through an integer round-trip for the
// addrspace casts (LDS aperture keeps the offset in the low 32 bits).
typedef int v4i_vs __attribute__((vector_size(16)));
#define GPTR(p) ((__attribute__((address_space(1))) v4i_vs*)(unsigned long long)(const void*)(p))
#define LPTR(p) ((__attribute__((address_space(3))) v4i_vs*)(unsigned int)(unsigned long long)(const void*)(p))
#define ASYNC_WAIT() asm volatile("s_wait_asynccnt 0x0" ::: "memory")
#endif

// ---------------- WMMA fragment helpers (contiguous k = 16*hi + j order, wave32) -------

// A fragment 16x32 f16 from row-major source (row stride ld halfs).
// lane L: m=L%16, hi=L/16; half j -> K=hi*16+j
__device__ __forceinline__ v16h load_a_frag(const _Float16* src, int ld) {
  const int l = threadIdx.x & 31;
  const int m = l & 15, hi = l >> 4;
  const _Float16* row = src + m * ld + hi * 16;
  v16h a;
#pragma unroll
  for (int j = 0; j < 8; ++j) a[j] = row[j];
#pragma unroll
  for (int j = 0; j < 8; ++j) a[8 + j] = row[8 + j];
  return a;
}

// B fragment 32x16: source laid out with column n contiguous along K
// (src[n*ld + k]).  lane L: n=L%16, half j -> K = (L/16)*16 + j
__device__ __forceinline__ v16h load_b_frag_kcol(const _Float16* src, int ld) {
  const int l = threadIdx.x & 31;
  const int n = l & 15, hi = l >> 4;
  const _Float16* col = src + n * ld + hi * 16;
  v16h b;
#pragma unroll
  for (int j = 0; j < 16; ++j) b[j] = col[j];
  return b;
}

// B fragment 32x16 from K-row-major source (src[k*ld + n])
__device__ __forceinline__ v16h load_b_frag_krow(const _Float16* src, int ld) {
  const int l = threadIdx.x & 31;
  const int n = l & 15, hi = l >> 4;
  v16h b;
#pragma unroll
  for (int j = 0; j < 16; ++j) b[j] = src[(hi * 16 + j) * ld + n];
  return b;
}

__device__ __forceinline__ v8f wmma32(v16h a, v16h b, v8f c) {
  return __builtin_amdgcn_wmma_f32_16x16x32_f16(false, a, false, b,
                                                (short)0, c, false, false);
}

// ---------------- Kernel 1: fused window-gather + QKV GEMM ------------------
// M=50176 tokens, N=2304, K=768.  Block tile 128x128; 8 waves in 4(M)x2(N)
// grid, each wave computes a 32x64 tile (8 WMMAs per K-step).
__global__ __launch_bounds__(256) void qkv_kernel(
    const float* __restrict__ x, const float* __restrict__ qkv_w,
    const float* __restrict__ qkv_b,
    _Float16* __restrict__ qb, _Float16* __restrict__ kb,
    _Float16* __restrict__ vb) {
  __shared__ _Float16 At[128 * 32];
  __shared__ _Float16 Bt[128 * 32];
  const int tid = threadIdx.x;
  const int bm = blockIdx.y, bn = blockIdx.x;

  // tile-fill mapping: 2 threads per row, 16 elements each
  const int ar = tid >> 1;
  const int acb = (tid & 1) * 16;

  // A-tile row -> windowed pixel of x
  const int g0 = bm * 128 + ar;
  const int win0 = g0 / NTOK, tok0 = g0 - win0 * NTOK;
  const int img = win0 >> 4, hy = (win0 >> 2) & 3, wx = win0 & 3;
  const int iy = tok0 / WSZ, ix = tok0 - iy * WSZ;
  const size_t xbase =
      (size_t)((img * 56 + hy * WSZ + iy) * 56 + wx * WSZ + ix) * CDIM;
  const size_t wbase = (size_t)(bn * 128 + ar) * CDIM;

  v8f zero = {};
  v8f acc[2][4];
#pragma unroll
  for (int mi = 0; mi < 2; ++mi)
#pragma unroll
    for (int nt = 0; nt < 4; ++nt) acc[mi][nt] = zero;

  const int wv = tid >> 5;
  const int wm = (wv & 3) * 32;   // wave M offset in block tile
  const int wn = (wv >> 2) * 64;  // wave N offset in block tile

  for (int k0 = 0; k0 < CDIM; k0 += 32) {
    __syncthreads();
#pragma unroll
    for (int j = 0; j < 16; ++j)
      At[ar * 32 + acb + j] = (_Float16)x[xbase + k0 + acb + j];
#pragma unroll
    for (int j = 0; j < 16; ++j)
      Bt[ar * 32 + acb + j] = (_Float16)qkv_w[wbase + k0 + acb + j];
    __syncthreads();
    v16h a0 = load_a_frag(At + wm * 32, 32);
    v16h a1 = load_a_frag(At + (wm + 16) * 32, 32);
    v16h b[4];
#pragma unroll
    for (int nt = 0; nt < 4; ++nt)
      b[nt] = load_b_frag_kcol(Bt + (wn + nt * 16) * 32, 32);
#pragma unroll
    for (int nt = 0; nt < 4; ++nt) {
      acc[0][nt] = wmma32(a0, b[nt], acc[0][nt]);
      acc[1][nt] = wmma32(a1, b[nt], acc[1][nt]);
    }
  }

  // epilogue: +bias, scatter to q/k/v [win][head][tok][hd] (f16)
  const int l = tid & 31, n = l & 15, hi = l >> 4;
#pragma unroll
  for (int nt = 0; nt < 4; ++nt) {
    const int o = bn * 128 + wn + nt * 16 + n;  // 0..2303
    const int three = o / CDIM;
    const int rem = o - three * CDIM;
    const int head = rem >> 6, d = rem & 63;
    _Float16* dst = (three == 0) ? qb : ((three == 1) ? kb : vb);
    const float bias = qkv_b[o];
#pragma unroll
    for (int mi = 0; mi < 2; ++mi)
#pragma unroll
      for (int r = 0; r < 8; ++r) {
        const int g = bm * 128 + wm + mi * 16 + r + 8 * hi;
        const int w2 = g / NTOK, t2 = g - w2 * NTOK;
        dst[(((size_t)w2 * NH + head) * NTOK + t2) * HD + d] =
            (_Float16)(acc[mi][nt][r] + bias);
      }
  }
}

// ---------------- Kernel 2: per-(window,head) attention ---------------------
// 128 threads = 4 waves.  170KB dynamic LDS (CDNA5 WGP has 320KB).
__global__ __launch_bounds__(128) void attn_kernel(
    const _Float16* __restrict__ qb, const _Float16* __restrict__ kb,
    const _Float16* __restrict__ vb, _Float16* __restrict__ ob) {
  extern __shared__ char smem[];
  _Float16* q_s = (_Float16*)smem;                  // 208x64 f16 = 26624 B
  _Float16* k_s = (_Float16*)(smem + 26624);        // 224x64 f16 = 28672 B
  _Float16* v_s = (_Float16*)(smem + 55296);        // 224x64 f16 = 28672 B
  float*    S_s = (float*)(smem + 83968);           // 4 x 16x224 f32 = 57344 B
  _Float16* P_s = (_Float16*)(smem + 141312);       // 4 x 16x224 f16 = 28672 B

  const int tid = threadIdx.x;
  const int bh = blockIdx.x;
  const int win = bh / NH, head = bh - win * NH;
  const size_t base = ((size_t)win * NH + head) * NTOK * HD;

#if HAVE_ASYNC_LDS
  // valid region: 196*64 halfs = 1568 b128 chunks per matrix, DMA'd to LDS
  for (int c = tid; c < (NTOK * HD) / 8; c += 128) {
    __builtin_amdgcn_global_load_async_to_lds_b128(GPTR(qb + base + c * 8),
                                                   LPTR(q_s + c * 8), 0, 0);
    __builtin_amdgcn_global_load_async_to_lds_b128(GPTR(kb + base + c * 8),
                                                   LPTR(k_s + c * 8), 0, 0);
    __builtin_amdgcn_global_load_async_to_lds_b128(GPTR(vb + base + c * 8),
                                                   LPTR(v_s + c * 8), 0, 0);
  }
  for (int i = NTOK * HD + tid; i < QPAD * HD; i += 128) q_s[i] = (_Float16)0.f;
  for (int i = NTOK * HD + tid; i < KPAD * HD; i += 128) {
    k_s[i] = (_Float16)0.f;
    v_s[i] = (_Float16)0.f;
  }
  ASYNC_WAIT();
#else
  for (int i = tid; i < QPAD * HD; i += 128)
    q_s[i] = (i < NTOK * HD) ? qb[base + i] : (_Float16)0.f;
  for (int i = tid; i < KPAD * HD; i += 128) {
    const bool ok = i < NTOK * HD;
    k_s[i] = ok ? kb[base + i] : (_Float16)0.f;
    v_s[i] = ok ? vb[base + i] : (_Float16)0.f;
  }
#endif
  __syncthreads();

  const int wv = tid >> 5, l = tid & 31;
  float*    Sw = S_s + wv * 16 * KPAD;
  _Float16* Pw = P_s + wv * 16 * KPAD;
  const int n = l & 15, hi = l >> 4;

  for (int qt = wv; qt < 13; qt += 4) {
    // S = (q @ k^T) * scale   (16 x 208 computed, 13 key tiles)
    v16h a0 = load_a_frag(q_s + qt * 16 * HD, HD);
    v16h a1 = load_a_frag(q_s + qt * 16 * HD + 32, HD);
    for (int kt = 0; kt < 13; ++kt) {
      v8f c = {};
      v16h b0 = load_b_frag_kcol(k_s + kt * 16 * HD, HD);
      v16h b1 = load_b_frag_kcol(k_s + kt * 16 * HD + 32, HD);
      c = wmma32(a0, b0, c);
      c = wmma32(a1, b1, c);
#pragma unroll
      for (int r = 0; r < 8; ++r)
        Sw[(r + 8 * hi) * KPAD + kt * 16 + n] = c[r] * 0.125f;  // 1/sqrt(64)
    }
    asm volatile("" ::: "memory");  // same-wave LDS ops stay in order

    // softmax over keys 0..195; lanes 0..15 own one row each
    if (l < 16) {
      float* row = Sw + l * KPAD;
      float mx = -3.0e38f;
      for (int j = 0; j < NTOK; ++j) mx = fmaxf(mx, row[j]);
      float s = 0.f;
      for (int j = 0; j < NTOK; ++j) {
        float e = __expf(row[j] - mx);
        row[j] = e;
        s += e;
      }
      const float inv = 1.f / s;
      _Float16* prow = Pw + l * KPAD;
      for (int j = 0; j < NTOK; ++j) prow[j] = (_Float16)(row[j] * inv);
      for (int j = NTOK; j < KPAD; ++j) prow[j] = (_Float16)0.f;  // mask pad
    }
    asm volatile("" ::: "memory");

    // O = P @ V   (16 x 64), K = 224 in 7 steps of 32
#pragma unroll
    for (int ht = 0; ht < 4; ++ht) {
      v8f c = {};
      for (int ks = 0; ks < 7; ++ks) {
        v16h a = load_a_frag(Pw + ks * 32, KPAD);
        v16h b = load_b_frag_krow(v_s + ks * 32 * HD + ht * 16, HD);
        c = wmma32(a, b, c);
      }
#pragma unroll
      for (int r = 0; r < 8; ++r) {
        const int tok = qt * 16 + r + 8 * hi;
        if (tok < NTOK)
          ob[((size_t)win * NTOK + tok) * CDIM + head * HD + ht * 16 + n] =
              (_Float16)c[r];
      }
    }
  }
}

// ---------------- Kernel 3: output projection + window scatter --------------
// Block tile 128x128, wave tile 32x64; A tile DMA'd (f16 passthrough).
__global__ __launch_bounds__(256) void proj_kernel(
    const _Float16* __restrict__ ob, const float* __restrict__ proj_w,
    const float* __restrict__ proj_b, float* __restrict__ out) {
  __shared__ _Float16 At[128 * 32];
  __shared__ _Float16 Bt[128 * 32];
  const int tid = threadIdx.x;
  const int bm = blockIdx.y, bn = blockIdx.x;

  const int ar = tid >> 1;
  const int acb = (tid & 1) * 16;
  const size_t abase = (size_t)(bm * 128 + ar) * CDIM;
  const size_t wbase = (size_t)(bn * 128 + ar) * CDIM;

  v8f zero = {};
  v8f acc[2][4];
#pragma unroll
  for (int mi = 0; mi < 2; ++mi)
#pragma unroll
    for (int nt = 0; nt < 4; ++nt) acc[mi][nt] = zero;

  const int wv = tid >> 5;
  const int wm = (wv & 3) * 32;
  const int wn = (wv >> 2) * 64;

  for (int k0 = 0; k0 < CDIM; k0 += 32) {
    __syncthreads();
#if HAVE_ASYNC_LDS
    __builtin_amdgcn_global_load_async_to_lds_b128(
        GPTR(ob + abase + k0 + acb), LPTR(At + ar * 32 + acb), 0, 0);
    __builtin_amdgcn_global_load_async_to_lds_b128(
        GPTR(ob + abase + k0 + acb + 8), LPTR(At + ar * 32 + acb + 8), 0, 0);
#else
#pragma unroll
    for (int j = 0; j < 16; ++j)
      At[ar * 32 + acb + j] = ob[abase + k0 + acb + j];
#endif
#pragma unroll
    for (int j = 0; j < 16; ++j)
      Bt[ar * 32 + acb + j] = (_Float16)proj_w[wbase + k0 + acb + j];
#if HAVE_ASYNC_LDS
    ASYNC_WAIT();
#endif
    __syncthreads();
    v16h a0 = load_a_frag(At + wm * 32, 32);
    v16h a1 = load_a_frag(At + (wm + 16) * 32, 32);
    v16h b[4];
#pragma unroll
    for (int nt = 0; nt < 4; ++nt)
      b[nt] = load_b_frag_kcol(Bt + (wn + nt * 16) * 32, 32);
#pragma unroll
    for (int nt = 0; nt < 4; ++nt) {
      acc[0][nt] = wmma32(a0, b[nt], acc[0][nt]);
      acc[1][nt] = wmma32(a1, b[nt], acc[1][nt]);
    }
  }

  const int l = tid & 31, n = l & 15, hi = l >> 4;
#pragma unroll
  for (int nt = 0; nt < 4; ++nt) {
    const int o = bn * 128 + wn + nt * 16 + n;  // output channel
    const float bias = proj_b[o];
#pragma unroll
    for (int mi = 0; mi < 2; ++mi)
#pragma unroll
      for (int r = 0; r < 8; ++r) {
        const int g = bm * 128 + wm + mi * 16 + r + 8 * hi;
        const int win = g / NTOK, tok = g - win * NTOK;
        const int img = win >> 4, hy = (win >> 2) & 3, wx = win & 3;
        const int iy = tok / WSZ, ix = tok - iy * WSZ;
        out[(size_t)((img * 56 + hy * WSZ + iy) * 56 + wx * WSZ + ix) * CDIM +
            o] = acc[mi][nt][r] + bias;
      }
  }
}

// ---------------- Host launcher ---------------------------------------------
extern "C" void kernel_launch(void* const* d_in, const int* in_sizes, int n_in,
                              void* d_out, int out_size, void* d_ws,
                              size_t ws_size, hipStream_t stream) {
  (void)in_sizes; (void)n_in; (void)out_size; (void)ws_size;
  const float* x      = (const float*)d_in[0];
  const float* qkv_w  = (const float*)d_in[1];
  const float* qkv_b  = (const float*)d_in[2];
  const float* proj_w = (const float*)d_in[3];
  const float* proj_b = (const float*)d_in[4];
  float* out = (float*)d_out;

  const size_t mat = (size_t)NWIN * NH * NTOK * HD;  // 38,535,168 halfs
  _Float16* qb = (_Float16*)d_ws;
  _Float16* kb = qb + mat;
  _Float16* vb = kb + mat;
  _Float16* ob = vb + mat;  // 50176 x 768 halfs (same element count as mat)

  const int ATTN_LDS = 169984;  // > 64KB: legal on CDNA5 (320KB WGP LDS)
  (void)hipFuncSetAttribute((const void*)attn_kernel,
                            hipFuncAttributeMaxDynamicSharedMemorySize,
                            ATTN_LDS);

  // QKV: M tiles 50176/128 = 392, N tiles 2304/128 = 18
  qkv_kernel<<<dim3(18, 392), 256, 0, stream>>>(x, qkv_w, qkv_b, qb, kb, vb);
  // Attention: one block per (window, head)
  attn_kernel<<<dim3(NWIN * NH), 128, ATTN_LDS, stream>>>(qb, kb, vb, ob);
  // Projection: N tiles 768/128 = 6
  proj_kernel<<<dim3(6, 392), 256, 0, stream>>>(ob, proj_w, proj_b, out);
}
